// LSTM_63024350102010
// MI455X (gfx1250) — compile-verified, matrix operand order checked by reading, not run on hardware
//
#include <hip/hip_runtime.h>

#define SIZE_XX 512
#define NS 1024          // size_short
#define NL 1024          // size_long
#define TT 4096
#define ZZ 1536
#define G4 4096          // 4 interleaved gate columns (f,i,g,rs per j)

typedef __attribute__((ext_vector_type(16))) __bf16 v16bf;
typedef __attribute__((ext_vector_type(8)))  float  v8f;
typedef __attribute__((ext_vector_type(8)))  unsigned short v8us;
typedef __attribute__((ext_vector_type(4)))  int    v4i;

#if defined(__has_builtin)
#if __has_builtin(__builtin_amdgcn_global_load_async_to_lds_b128)
#define HAVE_ASYNC_LDS 1
#endif
#endif

#define AS1 __attribute__((address_space(1)))
#define AS3 __attribute__((address_space(3)))

__device__ __forceinline__ unsigned short f2bf(float f) {
  unsigned u = __builtin_bit_cast(unsigned, f);
  unsigned rnd = 0x7fffu + ((u >> 16) & 1u);          // round-to-nearest-even
  return (unsigned short)((u + rnd) >> 16);
}
__device__ __forceinline__ float bf2f(unsigned short b) {
  return __builtin_bit_cast(float, (unsigned)b << 16);
}
__device__ __forceinline__ float sigf(float x) { return 1.f / (1.f + expf(-x)); }

// ---- async global->LDS staging helpers (16B per thread) -------------------
__device__ __forceinline__ void stage_b128(const unsigned short* src, unsigned short* dstLds) {
#ifdef HAVE_ASYNC_LDS
  __builtin_amdgcn_global_load_async_to_lds_b128(
      (v4i*)(const void*)src, (AS3 v4i*)dstLds, 0, 0);
#else
  *(v8us*)dstLds = *(const v8us*)src;
#endif
}
__device__ __forceinline__ void wait_async_le0() {
#ifdef HAVE_ASYNC_LDS
#if __has_builtin(__builtin_amdgcn_s_wait_asynccnt)
  __builtin_amdgcn_s_wait_asynccnt(0);
#else
  asm volatile("s_wait_asynccnt 0x0" ::: "memory");
#endif
#endif
}
__device__ __forceinline__ void wait_async_le1() {
#ifdef HAVE_ASYNC_LDS
#if __has_builtin(__builtin_amdgcn_s_wait_asynccnt)
  __builtin_amdgcn_s_wait_asynccnt(1);
#else
  asm volatile("s_wait_asynccnt 0x1" ::: "memory");
#endif
#endif
}

// ---------------------------------------------------------------------------
// Pack gate weights.
//   q = 4*j + s, s in {f,i,g,rs}
//   WxcT[q][r] (N-major / transposed, r<512 x-rows)  -> GEMM B operand
//   Whi [r][q] (K-major, h-rows)                     -> recurrent streaming
//   biasq[q] = bias_s[j]
// ---------------------------------------------------------------------------
__global__ void pack_gates(const float* __restrict__ wf, const float* __restrict__ wm,
                           const float* __restrict__ wt, const float* __restrict__ wrs,
                           const float* __restrict__ bf, const float* __restrict__ bm,
                           const float* __restrict__ bt, const float* __restrict__ brs,
                           unsigned short* __restrict__ Whi, unsigned short* __restrict__ WxcT,
                           float* __restrict__ biasq) {
  int idx = blockIdx.x * blockDim.x + threadIdx.x;     // over 1536*4096
  if (idx >= ZZ * G4) return;
  int r = idx >> 12;          // z row 0..1535
  int q = idx & (G4 - 1);
  int j = q >> 2, s = q & 3;
  const float* W = (s == 0) ? wf : (s == 1) ? wm : (s == 2) ? wt : wrs;
  float v = W[(size_t)r * NL + j];
  if (r < SIZE_XX) WxcT[(size_t)q * SIZE_XX + r] = f2bf(v);   // transposed
  else             Whi[(size_t)(r - SIZE_XX) * G4 + q] = f2bf(v);
  if (idx < G4) {
    const float* B = (s == 0) ? bf : (s == 1) ? bm : (s == 2) ? bt : brs;
    biasq[q] = B[j];
  }
}

// ---------------------------------------------------------------------------
// Convert wrt (K-major), wo (transposed -> N-major), x to bf16; zero state.
// ---------------------------------------------------------------------------
__global__ void pack_misc(const float* __restrict__ wrt, const float* __restrict__ wo,
                          const float* __restrict__ x,
                          unsigned short* __restrict__ wrtb, unsigned short* __restrict__ woT,
                          unsigned short* __restrict__ xb,
                          float* __restrict__ cvec, float* __restrict__ hvec,
                          unsigned* __restrict__ bar) {
  int idx = blockIdx.x * blockDim.x + threadIdx.x;     // over 4096*512 = 2,097,152
  if (idx < NL * NS) wrtb[idx] = f2bf(wrt[idx]);
  if (idx < NL * ZZ) {
    int r = idx / ZZ, n = idx % ZZ;
    woT[(size_t)n * NL + r] = f2bf(wo[idx]);           // transposed
  }
  if (idx < TT * SIZE_XX) xb[idx] = f2bf(x[idx]);
  if (idx < NL)           { cvec[idx] = 0.f; hvec[idx] = 0.f; }
  if (idx < 32)           bar[idx] = 0u;
}

// ---------------------------------------------------------------------------
// bf16 WMMA GEMM: C(MxN,f32) = A(MxK, K-major) * BT(NxK, K-major)^T
//   Block = 256 threads (8 waves) -> 128x64 output tile.
//   Wave w: rows [row0+16w,+16), 4 accumulators over 64 cols -> 4 WMMA/chunk
//   with A-fragment reuse. B chunks (64 n-rows x 32 K = 4 KB) are DOUBLE-
//   BUFFERED in LDS: chunk i+1's async global->LDS copy is issued before
//   computing chunk i, waiting only ASYNCcnt<=1 (in-order retirement) so the
//   copy overlaps the WMMAs. Fragments read back as 2x ds_load_b128 per lane
//   (80 B padded rows, conflict-free).
//   mode 0: C = acc + bias[n];  mode 1: C = sigmoid(acc + bias[n])
// ---------------------------------------------------------------------------
#define SB_STRIDE 40   // ushorts per LDS n-row (80 B: 16B-aligned, bank-padded)

__global__ void gemm_bf16_wmma(const unsigned short* __restrict__ A,
                               const unsigned short* __restrict__ BT,
                               float* __restrict__ C,
                               const float* __restrict__ bias,
                               int M, int N, int K, int mode) {
  __shared__ unsigned short sB[2][64 * SB_STRIDE];
  const int tilesN = N >> 6;
  const int row0 = (blockIdx.x / tilesN) << 7;
  const int col0 = (blockIdx.x % tilesN) << 6;
  const int wave = threadIdx.x >> 5;
  const int lane = threadIdx.x & 31;
  const int mn = lane & 15;
  const int lh = lane >> 4;
  const int lh8 = lh << 3;
  const int sn = threadIdx.x >> 2;        // staging: n-row 0..63
  const int sc = (threadIdx.x & 3) << 3;  // staging: K sub-chunk (ushorts)

  v8f acc[4];
#pragma unroll
  for (int c = 0; c < 4; ++c)
#pragma unroll
    for (int i = 0; i < 8; ++i) acc[c][i] = 0.f;

  const unsigned short* Arow = A + (size_t)(row0 + (wave << 4) + mn) * K;
  const unsigned short* Bstage = BT + (size_t)(col0 + sn) * K + sc;
  unsigned short* sDst = &sB[0][sn * SB_STRIDE + sc];
  const int sBuf = 64 * SB_STRIDE;        // ushort stride between buffers

  stage_b128(Bstage, sDst);               // prologue: chunk 0 -> buffer 0

  int ib = 0;
  for (int k0 = 0; k0 < K; k0 += 32, ib ^= 1) {
    if (k0 + 32 < K) {
      __syncthreads();                    // prev iter's readers of buf[ib^1] done
      stage_b128(Bstage + k0 + 32, sDst + (ib ^ 1) * sBuf);
      wait_async_le1();                   // chunk k0's (older) copy complete
    } else {
      wait_async_le0();
    }
    __syncthreads();                      // B chunk visible to all waves

    // A fragment: per-lane K runs [lh*8,+8) and [16+lh*8,+8) -> 2x b128
    v8us a0 = *(const v8us*)(Arow + k0 + lh8);
    v8us a1 = *(const v8us*)(Arow + k0 + 16 + lh8);
    v16bf a;
#pragma unroll
    for (int e = 0; e < 8; ++e) {
      a[e]     = __builtin_bit_cast(__bf16, a0[e]);
      a[e + 8] = __builtin_bit_cast(__bf16, a1[e]);
    }
#pragma unroll
    for (int c = 0; c < 4; ++c) {
      const unsigned short* bp = &sB[ib][((c << 4) + mn) * SB_STRIDE];
      v8us b0 = *(const v8us*)(bp + lh8);
      v8us b1 = *(const v8us*)(bp + 16 + lh8);
      v16bf b;
#pragma unroll
      for (int e = 0; e < 8; ++e) {
        b[e]     = __builtin_bit_cast(__bf16, b0[e]);
        b[e + 8] = __builtin_bit_cast(__bf16, b1[e]);
      }
      acc[c] = __builtin_amdgcn_wmma_f32_16x16x32_bf16(false, a, false, b,
                                                       (short)0, acc[c], false, false);
    }
  }

#pragma unroll
  for (int c = 0; c < 4; ++c) {
    int coln = col0 + (c << 4) + mn;
    float bv = bias[coln];
#pragma unroll
    for (int v = 0; v < 8; ++v) {
      int m = row0 + (wave << 4) + v + (lh << 3);
      float val = acc[c][v] + bv;
      if (mode) val = sigf(val);
      C[(size_t)m * N + coln] = val;
    }
  }
}

// ---------------------------------------------------------------------------
// Grid-wide barrier for the persistent recurrent kernel (16 workgroups).
// ---------------------------------------------------------------------------
__device__ __forceinline__ void grid_barrier(unsigned* arrive, unsigned* gen, unsigned ngrp) {
  __threadfence();
  __syncthreads();
  if (threadIdx.x == 0) {
    unsigned g = __hip_atomic_load(gen, __ATOMIC_ACQUIRE, __HIP_MEMORY_SCOPE_AGENT);
    unsigned a = __hip_atomic_fetch_add(arrive, 1u, __ATOMIC_ACQ_REL, __HIP_MEMORY_SCOPE_AGENT);
    if (a + 1 == ngrp) {
      __hip_atomic_store(arrive, 0u, __ATOMIC_RELAXED, __HIP_MEMORY_SCOPE_AGENT);
      __hip_atomic_fetch_add(gen, 1u, __ATOMIC_RELEASE, __HIP_MEMORY_SCOPE_AGENT);
    } else {
      while (__hip_atomic_load(gen, __ATOMIC_ACQUIRE, __HIP_MEMORY_SCOPE_AGENT) == g)
        __builtin_amdgcn_s_sleep(2);
    }
  }
  __syncthreads();
}

// ---------------------------------------------------------------------------
// Sequential core. 16 blocks x 256 threads; thread q=4j+s owns gate s of
// column j. gates = Xproj[t,q] + h . Whi[:,q]; quad shfl -> c_j; barrier;
// rt_j = tanh(c . wrt[:,j]) 4-way split + shfl reduce; h stored fp32 + bf16.
// ---------------------------------------------------------------------------
#define RBLK 16
#define RTHR 256

__global__ void recurrent_kernel(const float* __restrict__ Xproj,
                                 const unsigned short* __restrict__ Whi,
                                 const unsigned short* __restrict__ Wrt,
                                 const float* __restrict__ brt,
                                 float* __restrict__ cvec, float* __restrict__ hvec,
                                 unsigned short* __restrict__ Hbf,
                                 unsigned* __restrict__ bar) {
  __shared__ float sh[NS];
  __shared__ float sc[NL];
  const int tid = threadIdx.x;
  const int q = blockIdx.x * RTHR + tid;
  const int j = q >> 2, s = q & 3;
  const int lane = tid & 31;
  const int base = lane & ~3;
  unsigned* arrive = bar;
  unsigned* gen = bar + 16;        // separate cacheline
  float creg = 0.f;                // c_j replicated across the quad

  for (int t = 0; t < TT; ++t) {
    for (int r = tid; r < NS; r += RTHR) sh[r] = hvec[r];
    __syncthreads();
    __builtin_prefetch(&Xproj[(size_t)(t + 1) * G4 + q], 0, 1);
    float acc = Xproj[(size_t)t * G4 + q];       // x-projection + bias (precomputed)
#pragma unroll 8
    for (int r = 0; r < NS; ++r)
      acc += sh[r] * bf2f(Whi[(size_t)r * G4 + q]);
    float val = (s == 2) ? tanhf(acc) : sigf(acc);
    float f  = __shfl(val, base + 0, 32);
    float iv = __shfl(val, base + 1, 32);
    float g  = __shfl(val, base + 2, 32);
    float rs = __shfl(val, base + 3, 32);
    creg = creg * f + iv * g;
    if (s == 0) cvec[j] = creg;
    grid_barrier(arrive, gen, RBLK);             // c complete everywhere

    for (int r = tid; r < NL; r += RTHR) sc[r] = cvec[r];
    __syncthreads();
    float p = 0.f;
#pragma unroll 8
    for (int r = s * 256; r < s * 256 + 256; ++r)
      p += sc[r] * bf2f(Wrt[(size_t)r * NS + j]);
    p += __shfl_xor(p, 1, 32);
    p += __shfl_xor(p, 2, 32);
    float hj = rs * tanhf(p + brt[j]);
    if (s == 0) hvec[j] = hj;
    if (s == 1) Hbf[(size_t)t * NS + j] = f2bf(hj);
    grid_barrier(arrive, gen, RBLK);             // h complete everywhere
  }
}

// ---------------------------------------------------------------------------
extern "C" void kernel_launch(void* const* d_in, const int* in_sizes, int n_in,
                              void* d_out, int out_size, void* d_ws, size_t ws_size,
                              hipStream_t stream) {
  (void)in_sizes; (void)n_in; (void)out_size; (void)ws_size;
  const float* x   = (const float*)d_in[0];
  const float* wf  = (const float*)d_in[1];
  const float* bf  = (const float*)d_in[2];
  const float* wm  = (const float*)d_in[3];
  const float* bm  = (const float*)d_in[4];
  const float* wt  = (const float*)d_in[5];
  const float* bt  = (const float*)d_in[6];
  const float* wrs = (const float*)d_in[7];
  const float* brs = (const float*)d_in[8];
  const float* wrt = (const float*)d_in[9];
  const float* brt = (const float*)d_in[10];
  const float* wo  = (const float*)d_in[11];
  const float* bo  = (const float*)d_in[12];

  char* w = (char*)d_ws;
  size_t off = 0;
  auto carve = [&](size_t bytes) { char* p = w + off; off = (off + bytes + 255) & ~(size_t)255; return p; };
  unsigned short* Whi   = (unsigned short*)carve((size_t)NL * G4 * 2);      //  8 MB
  unsigned short* WxcT  = (unsigned short*)carve((size_t)G4 * SIZE_XX * 2); //  4 MB (transposed)
  unsigned short* Wrtb  = (unsigned short*)carve((size_t)NL * NS * 2);      //  2 MB
  unsigned short* WoT   = (unsigned short*)carve((size_t)ZZ * NL * 2);      //  3 MB (transposed)
  unsigned short* Xbf   = (unsigned short*)carve((size_t)TT * SIZE_XX * 2); //  4 MB
  float*          biasq = (float*)carve((size_t)G4 * 4);
  float*          cvec  = (float*)carve((size_t)NL * 4);
  float*          hvec  = (float*)carve((size_t)NL * 4);
  unsigned*       bar   = (unsigned*)carve(32 * 4);
  float*          Xproj = (float*)carve((size_t)TT * G4 * 4);               // 64 MB
  unsigned short* Hbf   = (unsigned short*)carve((size_t)TT * NS * 2);      //  8 MB

  // Phase 0: weight packing / bf16 conversion / state+barrier init.
  pack_gates<<<(ZZ * G4 + 255) / 256, 256, 0, stream>>>(wf, wm, wt, wrs, bf, bm, bt, brs,
                                                        Whi, WxcT, biasq);
  pack_misc<<<(TT * SIZE_XX + 255) / 256, 256, 0, stream>>>(wrt, wo, x, Wrtb, WoT,
                                                            Xbf, cvec, hvec, bar);

  // Phase 1: time-parallel input projection  Xproj = Xbf @ WxcT^T + biasq  (WMMA)
  gemm_bf16_wmma<<<(TT / 128) * (G4 / 64), 256, 0, stream>>>(Xbf, WxcT, Xproj, biasq,
                                                             TT, G4, SIZE_XX, 0);

  // Phase 2: sequential recurrence (persistent, grid barriers).
  recurrent_kernel<<<RBLK, RTHR, 0, stream>>>(Xproj, Whi, Wrtb, brt, cvec, hvec, Hbf, bar);

  // Phase 3: deferred output layer  out = sigmoid(Hbf @ WoT^T + bo)  (WMMA)
  gemm_bf16_wmma<<<(TT / 128) * (ZZ / 64), 256, 0, stream>>>(Hbf, WoT, (float*)d_out, bo,
                                                             TT, ZZ, NL, 1);
}
